// GATNet_MLP_33930241638750
// MI455X (gfx1250) — compile-verified, matrix operand order checked by reading, not run on hardware
//
#include <hip/hip_runtime.h>
#include <math.h>

// ---------------------------------------------------------------------------
// GATNet (3x GATConv w/ edge attrs) + MLP head + global mean pool, MI455X.
//
// Roofline: dense GEMMs ~25 GFLOP (negligible vs WMMA peak); edge
// gather/scatter ~5 GB of HBM traffic dominates (~220us @ 23.3 TB/s).
// => GEMMs via v_wmma_f32_16x16x32_f16 (fp16 operands, fp32 accum) with
//    LDS-staged B tiles (8-way reuse across waves), graph part via lean
//    vectorized gather + fp32 atomic scatter.
// ---------------------------------------------------------------------------

#define DIN   128
#define FW    256      // feature width of every hidden representation
#define EDIM  5
#define NGR   64
#define DOUT  32

#define BSTRIDE 40     // LDS halves per B column (pad: bank-conflict-free b128)

typedef _Float16 v16h __attribute__((ext_vector_type(16)));
typedef float    v8f  __attribute__((ext_vector_type(8)));

__device__ __forceinline__ float gelu_exact(float x) {
    return 0.5f * x * (1.0f + erff(x * 0.70710678118654752440f));
}

// ---------------------------------------------------------------------------
// WMMA GEMM: C[M,Ncol] = act(A[M,K] @ W[K,Ncol] + bias)
// Block = 8 waves, tile = 128 rows x 32 cols. Per K-chunk (32):
//   - 256 threads cooperatively stage B[32k x 32n] into LDS as fp16,
//     transposed ([col][k], stride 40 halves) so each lane's fragment is
//     two contiguous 16B ds reads, conflict-free across 16 lanes.
//   - each wave loads one A fragment from global (fp32->fp16) and issues
//     TWO wmma ops (col subtiles 0 and 1) sharing that A fragment.
// ACT: 0 = linear (+bias if non-null), 1 = bias + exact GELU.
// K % 32 == 0, Ncol % 32 == 0.
// ---------------------------------------------------------------------------
template<int ACT>
__global__ __launch_bounds__(256)
void gemm_wmma(const float* __restrict__ A, const float* __restrict__ W,
               const float* __restrict__ bias, float* __restrict__ C,
               int M, int K, int Ncol)
{
    __shared__ _Float16 Bs[32 * BSTRIDE];          // 2.5 KB

    const int t    = threadIdx.x;
    const int lane = t & 31;
    const int wave = t >> 5;
    const int row0 = blockIdx.x * 128 + wave * 16;
    const int col0 = blockIdx.y * 32;

    const int half = lane >> 4;                    // 0: lanes 0-15, 1: 16-31
    const int id15 = lane & 15;
    int m = row0 + id15; if (m >= M) m = M - 1;    // clamp: EXEC stays all-1s
    const int kb = half * 8;                       // fragment K base per half

    // cooperative B staging indices: thread t -> k row bk, 4 cols at bc4
    const int bk  = t >> 3;                        // 0..31
    const int bc4 = (t & 7) * 4;                   // 0,4,..,28

    v8f acc0 = {}, acc1 = {};
    for (int k0 = 0; k0 < K; k0 += 32) {
        const float4 wv = *(const float4*)(W + (size_t)(k0 + bk) * Ncol + col0 + bc4);
        __syncthreads();                           // LDS reads of prev chunk done
        Bs[(bc4 + 0) * BSTRIDE + bk] = (_Float16)wv.x;
        Bs[(bc4 + 1) * BSTRIDE + bk] = (_Float16)wv.y;
        Bs[(bc4 + 2) * BSTRIDE + bk] = (_Float16)wv.z;
        Bs[(bc4 + 3) * BSTRIDE + bk] = (_Float16)wv.w;

        // A fragment: lane half holds K = {kb..kb+7} U {kb+16..kb+23}
        v16h a;
        const float* Ar = A + (size_t)m * K + (k0 + kb);
        #pragma unroll
        for (int i = 0; i < 8; ++i) a[i]     = (_Float16)Ar[i];
        #pragma unroll
        for (int i = 0; i < 8; ++i) a[8 + i] = (_Float16)Ar[16 + i];

        __syncthreads();                           // B tile visible

        v16h b0, b1;
        const _Float16* B0 = &Bs[id15 * BSTRIDE + kb];          // cols 0..15
        const _Float16* B1 = &Bs[(16 + id15) * BSTRIDE + kb];   // cols 16..31
        #pragma unroll
        for (int i = 0; i < 8; ++i) { b0[i] = B0[i]; b0[8 + i] = B0[16 + i]; }
        #pragma unroll
        for (int i = 0; i < 8; ++i) { b1[i] = B1[i]; b1[8 + i] = B1[16 + i]; }

        acc0 = __builtin_amdgcn_wmma_f32_16x16x32_f16(
            false, a, false, b0, (short)0, acc0, false, false);
        acc1 = __builtin_amdgcn_wmma_f32_16x16x32_f16(
            false, a, false, b1, (short)0, acc1, false, false);
    }

    const int  c0  = col0 + id15;
    const int  c1  = col0 + 16 + id15;
    const float bv0 = bias ? bias[c0] : 0.0f;
    const float bv1 = bias ? bias[c1] : 0.0f;
    #pragma unroll
    for (int v = 0; v < 8; ++v) {
        int r = row0 + v + half * 8;               // D layout: VGPR v -> M=v (+8 hi)
        if (r < M) {
            float v0 = acc0[v] + bv0;
            float v1 = acc1[v] + bv1;
            if (ACT == 1) { v0 = gelu_exact(v0); v1 = gelu_exact(v1); }
            C[(size_t)r * Ncol + c0] = v0;
            C[(size_t)r * Ncol + c1] = v1;
        }
    }
}

// ---------------------------------------------------------------------------
// Per-node attention scores: a_s[i,h] = <h[i,h,:], att_s[h,:]>, same for a_d.
// ---------------------------------------------------------------------------
__global__ void node_scores(const float* __restrict__ h,
                            const float* __restrict__ att_s,
                            const float* __restrict__ att_d,
                            float* __restrict__ as_o, float* __restrict__ ad_o,
                            int n, int nh, int C)
{
    int idx = blockIdx.x * blockDim.x + threadIdx.x;
    if (idx >= n * nh) return;
    int i = idx / nh, hh = idx - i * nh;
    const float* hr = h + (size_t)i * FW + hh * C;
    const float* s  = att_s + hh * C;
    const float* d  = att_d + hh * C;
    float accs = 0.f, accd = 0.f;
    for (int c = 0; c < C; ++c) { float v = hr[c]; accs += v * s[c]; accd += v * d[c]; }
    as_o[idx] = accs;
    ad_o[idx] = accd;
}

// order-preserving float <-> uint key (for atomic segment-max)
__device__ __forceinline__ unsigned f2key(float f) {
    unsigned b = __float_as_uint(f);
    return b ^ ((b >> 31) ? 0xFFFFFFFFu : 0x80000000u);
}
__device__ __forceinline__ float key2f(unsigned k) {
    unsigned b = (k & 0x80000000u) ? (k ^ 0x80000000u) : ~k;
    return __uint_as_float(b);
}

// ---------------------------------------------------------------------------
// alpha[e,h] = leaky_relu(a_s[src]+a_d[dst]+ ea[e,:]@Me[:,h]) ; segment max.
// Me[d][h] = <We[d, h*C:(h+1)*C], ae[h,:]> — tiny, built in LDS per block.
// ---------------------------------------------------------------------------
__global__ void edge_alpha(const int* __restrict__ src, const int* __restrict__ dst,
                           const float* __restrict__ ea,
                           const float* __restrict__ We, const float* __restrict__ ae,
                           const float* __restrict__ as_, const float* __restrict__ ad_,
                           float* __restrict__ alpha_o, unsigned* __restrict__ mkeys,
                           int E, int nh, int C)
{
    __shared__ float Me[EDIM * 8];
    int t = threadIdx.x;
    if (t < EDIM * nh) {
        int d = t / nh, hh = t - d * nh;
        const float* wr = We + (size_t)d * FW + hh * C;
        const float* ar = ae + hh * C;
        float acc = 0.f;
        for (int c = 0; c < C; ++c) acc += wr[c] * ar[c];
        Me[d * nh + hh] = acc;
    }
    __syncthreads();

    int idx = blockIdx.x * blockDim.x + t;
    if (idx >= E * nh) return;
    int e = idx / nh, hh = idx - e * nh;
    int s = src[e], dd = dst[e];
    float a = as_[s * nh + hh] + ad_[dd * nh + hh];
    #pragma unroll
    for (int d = 0; d < EDIM; ++d) a += ea[(size_t)e * EDIM + d] * Me[d * nh + hh];
    a = (a > 0.f) ? a : 0.2f * a;          // leaky_relu, slope 0.2
    alpha_o[idx] = a;
    atomicMax(&mkeys[dd * nh + hh], f2key(a));
}

// p = exp(alpha - max[dst]) ; denom[dst] += p
__global__ void edge_exp(const int* __restrict__ dst, const unsigned* __restrict__ mkeys,
                         float* __restrict__ p, float* __restrict__ denom, int E, int nh)
{
    int idx = blockIdx.x * blockDim.x + threadIdx.x;
    if (idx >= E * nh) return;
    int e = idx / nh, hh = idx - e * nh;
    int dd = dst[e];
    float m  = key2f(mkeys[dd * nh + hh]);
    float pv = __expf(p[idx] - m);
    p[idx] = pv;
    atomicAdd(&denom[dd * nh + hh], pv);
}

// out[dst,:] += p[e,h(c)] * h[src,:]  — 64 lanes/edge, 4 channels each
__global__ void edge_aggregate(const int* __restrict__ src, const int* __restrict__ dst,
                               const float* __restrict__ h, const float* __restrict__ p,
                               float* __restrict__ out, int E, int nh)
{
    int idx = blockIdx.x * blockDim.x + threadIdx.x;
    if (idx >= E * 64) return;
    int e  = idx >> 6;
    int c4 = (idx & 63) * 4;
    int hh = (nh == 1) ? 0 : (c4 >> 5);
    int s  = src[e], dd = dst[e];
    if ((idx & 63) == 0 && e + 8 < E)      // stream src rows ahead of the gather
        __builtin_prefetch(h + (size_t)src[e + 8] * FW, 0, 0);
    float w = p[e * nh + hh];
    const float4 v = *(const float4*)(h + (size_t)s * FW + c4);
    float* o = out + (size_t)dd * FW + c4;
    atomicAdd(o + 0, v.x * w);
    atomicAdd(o + 1, v.y * w);
    atomicAdd(o + 2, v.z * w);
    atomicAdd(o + 3, v.w * w);
}

// out = elu(out / (denom + 1e-16) + bias)
__global__ void finalize_elu(float* __restrict__ out, const float* __restrict__ denom,
                             const float* __restrict__ bias, int n, int nh)
{
    int idx = blockIdx.x * blockDim.x + threadIdx.x;
    if (idx >= n * FW) return;
    int i = idx >> 8, c = idx & 255;
    int hh = (nh == 1) ? 0 : (c >> 5);
    float v = out[idx] / (denom[i * nh + hh] + 1e-16f) + bias[c];
    out[idx] = (v > 0.f) ? v : (__expf(v) - 1.0f);
}

// global mean pool: atomic sums + counts, then divide
__global__ void pool_sum(const float* __restrict__ h5, const int* __restrict__ batch,
                         float* __restrict__ sums, float* __restrict__ cnt, int n)
{
    int idx = blockIdx.x * blockDim.x + threadIdx.x;
    if (idx >= n * DOUT) return;
    int i = idx >> 5, c = idx & 31;
    int g = batch[i];
    atomicAdd(&sums[g * DOUT + c], h5[idx]);
    if (c == 0) atomicAdd(&cnt[g], 1.0f);
}

__global__ void pool_div(const float* __restrict__ sums, const float* __restrict__ cnt,
                         float* __restrict__ out)
{
    int idx = blockIdx.x * blockDim.x + threadIdx.x;
    if (idx >= NGR * DOUT) return;
    out[idx] = sums[idx] / fmaxf(cnt[idx >> 5], 1.0f);
}

// ---------------------------------------------------------------------------
extern "C" void kernel_launch(void* const* d_in, const int* in_sizes, int n_in,
                              void* d_out, int out_size, void* d_ws, size_t ws_size,
                              hipStream_t stream)
{
    const float* x    = (const float*)d_in[0];
    const int*   ei   = (const int*)  d_in[1];
    const float* ea   = (const float*)d_in[2];
    const int*   batch= (const int*)  d_in[3];
    const float* W1   = (const float*)d_in[4];
    const float* as1  = (const float*)d_in[5];
    const float* ad1  = (const float*)d_in[6];
    const float* We1  = (const float*)d_in[7];
    const float* ae1  = (const float*)d_in[8];
    const float* b1   = (const float*)d_in[9];
    const float* W2   = (const float*)d_in[10];
    const float* as2  = (const float*)d_in[11];
    const float* ad2  = (const float*)d_in[12];
    const float* We2  = (const float*)d_in[13];
    const float* ae2  = (const float*)d_in[14];
    const float* b2   = (const float*)d_in[15];
    const float* W3   = (const float*)d_in[16];
    const float* as3  = (const float*)d_in[17];
    const float* ad3  = (const float*)d_in[18];
    const float* We3  = (const float*)d_in[19];
    const float* ae3  = (const float*)d_in[20];
    const float* b3   = (const float*)d_in[21];
    const float* fcW1 = (const float*)d_in[22];
    const float* fcb1 = (const float*)d_in[23];
    const float* fcW2 = (const float*)d_in[24];
    const float* fcb2 = (const float*)d_in[25];

    const int N = in_sizes[0] / DIN;
    const int E = in_sizes[1] / 2;
    const int* srcI = ei;
    const int* dstI = ei + E;

    // ---- workspace layout (~122 MB) ----
    char* ws = (char*)d_ws;
    size_t off = 0;
    auto carve = [&](size_t bytes) -> void* {
        void* pp = ws + off;
        off = (off + bytes + 255) & ~(size_t)255;
        return pp;
    };
    float*    bufA  = (float*)   carve((size_t)N * FW * 4);   // h_lin of layer
    float*    bufB  = (float*)   carve((size_t)N * FW * 4);   // aggregated out
    float*    asb   = (float*)   carve((size_t)N * 8 * 4);
    float*    adb   = (float*)   carve((size_t)N * 8 * 4);
    unsigned* mkeys = (unsigned*)carve((size_t)N * 8 * 4);
    float*    denom = (float*)   carve((size_t)N * 8 * 4);
    float*    pbuf  = (float*)   carve((size_t)E * 8 * 4);    // alpha -> p ; later h5
    float*    sums  = (float*)   carve((size_t)NGR * DOUT * 4);
    float*    cnt   = (float*)   carve((size_t)NGR * 4);

    auto run_layer = [&](const float* X, int K, const float* W,
                         const float* atts, const float* attd,
                         const float* We, const float* aee,
                         const float* bb, int nh) {
        const int C = FW / nh;
        dim3 gg((N + 127) / 128, FW / 32);
        gemm_wmma<0><<<gg, 256, 0, stream>>>(X, W, nullptr, bufA, N, K, FW);
        (void)hipMemsetAsync(mkeys, 0, (size_t)N * nh * 4, stream);
        (void)hipMemsetAsync(denom, 0, (size_t)N * nh * 4, stream);
        (void)hipMemsetAsync(bufB,  0, (size_t)N * FW * 4, stream);
        int t1 = N * nh;
        node_scores<<<(t1 + 255) / 256, 256, 0, stream>>>(bufA, atts, attd, asb, adb, N, nh, C);
        int t2 = E * nh;
        edge_alpha<<<(t2 + 255) / 256, 256, 0, stream>>>(srcI, dstI, ea, We, aee,
                                                         asb, adb, pbuf, mkeys, E, nh, C);
        edge_exp<<<(t2 + 255) / 256, 256, 0, stream>>>(dstI, mkeys, pbuf, denom, E, nh);
        long long t3 = (long long)E * 64;
        edge_aggregate<<<(unsigned)((t3 + 255) / 256), 256, 0, stream>>>(srcI, dstI, bufA,
                                                                        pbuf, bufB, E, nh);
        int t4 = N * FW;
        finalize_elu<<<(t4 + 255) / 256, 256, 0, stream>>>(bufB, denom, bb, N, nh);
    };

    run_layer(x,    DIN, W1, as1, ad1, We1, ae1, b1, 8);
    run_layer(bufB, FW,  W2, as2, ad2, We2, ae2, b2, 8);
    run_layer(bufB, FW,  W3, as3, ad3, We3, ae3, b3, 1);

    // MLP head: gelu(h @ fcW1 + fcb1) @ fcW2 + fcb2
    gemm_wmma<1><<<dim3((N + 127) / 128, FW / 32), 256, 0, stream>>>(
        bufB, fcW1, fcb1, bufA, N, FW, FW);
    float* h5 = pbuf;   // reuse (E*8*4 >= N*32*4)
    gemm_wmma<0><<<dim3((N + 127) / 128, DOUT / 32), 256, 0, stream>>>(
        bufA, fcW2, fcb2, h5, N, FW, DOUT);

    // global mean pool
    (void)hipMemsetAsync(sums, 0, (size_t)NGR * DOUT * 4, stream);
    (void)hipMemsetAsync(cnt,  0, (size_t)NGR * 4, stream);
    int t5 = N * DOUT;
    pool_sum<<<(t5 + 255) / 256, 256, 0, stream>>>(h5, batch, sums, cnt, N);
    pool_div<<<(NGR * DOUT + 255) / 256, 256, 0, stream>>>(sums, cnt, (float*)d_out);
}
